// LearnableTensorProduct_63720134804197
// MI455X (gfx1250) — matrix operand order repaired and checked by reading, not compiled, not run on hardware
//
#include <hip/hip_runtime.h>

// ---------------- types ----------------
typedef __attribute__((ext_vector_type(16))) __bf16 v16bf;
typedef __attribute__((ext_vector_type(8)))  __bf16 v8bf;
typedef __attribute__((ext_vector_type(8)))  float  v8f;

// ---------------- problem constants ----------------
constexpr int Z        = 16384;
constexpr int ZT       = 16;      // z-tile per block
constexpr int MUL_MID  = 64;
constexpr int DIM_MID  = 9;
constexpr int DIM_OUT  = 2304;

// GEMM K sizes: K_L = 64 * NL,  NL = {3,6,6}
constexpr int K0 = 192, K12 = 384;

// LDS X-tile layout (bf16 element offsets). Row strides padded so that the 16
// A-fragment lanes (rows) start in distinct 4-dword bank groups for ds_load_b128.
constexpr int ST0  = K0  + 24;                 // 216 elems (108 dw, 108%64=44=4*11)
constexpr int ST12 = K12 + 24;                 // 408 elems (204 dw, 204%64=12=4*3)
constexpr int X0_BASE = 0;                     // 16 rows  (L=0, rows = z_local)
constexpr int X1_BASE = 16 * ST0;              // 3456 ; 48 rows (L=1, row = m*16+zl)
constexpr int X2_BASE = X1_BASE + 48 * ST12;   // 23040; 80 rows (L=2)
constexpr int LDS_ELEMS = X2_BASE + 80 * ST12; // 55680
constexpr size_t LDS_BYTES = (size_t)LDS_ELEMS * 2;  // 111360 B

// bf16 W staging offsets inside d_ws (elements)
constexpr int WB0_OFF = 0;
constexpr int WB1_OFF = 256 * K0;              // 49152
constexpr int WB2_OFF = WB1_OFF + 256 * K12;   // 147456
constexpr int WB_TOTAL = WB2_OFF + 256 * K12;  // 245760 elems = 491520 B

// norms: 1/sqrt(64*NL)
#define NORM_L0 0.07216878364870323f   // 1/sqrt(192)
#define NORM_L12 0.05103103630798288f  // 1/sqrt(384)

#if defined(__has_builtin)
#if __has_builtin(__builtin_amdgcn_sched_barrier)
#define SCHED_FENCE() __builtin_amdgcn_sched_barrier(0)
#endif
#endif
#ifndef SCHED_FENCE
#define SCHED_FENCE()
#endif

// ---------------- compile-time CG structural-zero predicate -------------------
// Real-basis Clebsch-Gordan C[m3; m1, m2] (component indices a=m1+l1 etc.) is
// nonzero only if:
//  (1) |m3| == |m1|+|m2|  or  |m3| == ||m1|-|m2||   (complex CG: m3 = m1+m2,
//      real basis mixes only +/-m)
//  (2) sin/cos class: (m3<0) == (m1<0) ^ (m2<0) ^ ((l1+l2+lo) odd)
//      (odd couplings are the antisymmetric ones, e.g. the eps-tensor 1x1->1)
// Coefficient VALUES are still read from the runtime mixing matrix; this only
// prunes terms that are exactly zero by symmetry.
constexpr int cabs(int v) { return v < 0 ? -v : v; }
constexpr bool cg_nonzero(int l1, int l2, int lo, int a, int b, int m) {
  const int m1 = a - l1, m2 = b - l2, m3 = m - lo;
  const int am1 = cabs(m1), am2 = cabs(m2), am3 = cabs(m3);
  if (am3 != am1 + am2 && am3 != cabs(am1 - am2)) return false;
  const bool s1 = m1 < 0, s2 = m2 < 0, s3 = m3 < 0;
  const bool odd = ((l1 + l2 + lo) & 1) != 0;
  return s3 == ((s1 != s2) != odd);
}

// ---------------- TP helpers (fully unrolled; coeffs via scalar loads) ----------------
template<int l1, int l2, int IOFF, int JOFF>
__device__ __forceinline__ void tp_prod(const float (&x1v)[9], const float (&x2v)[9],
                                        float (&p)[(2*l1+1)*(2*l2+1)]) {
#pragma unroll
  for (int a = 0; a < 2*l1+1; ++a)
#pragma unroll
    for (int b = 0; b < 2*l2+1; ++b)
      p[a*(2*l2+1)+b] = x1v[IOFF+a] * x2v[JOFF+b];
}

// One (l1,l2,lo) CG block: tp[DST+m] = sum_{a,b} M[KG+m, IOFF+a, JOFF+b] * p[a,b]
// All M offsets are compile-time -> uniform scalar loads through the K$ path.
// cg_nonzero folds after unrolling, so structurally-zero terms emit nothing.
template<int l1, int l2, int lo, int IOFF, int JOFF, int KG, int DST>
__device__ __forceinline__ void tp_lo(const float* __restrict__ mx,
                                      const float (&p)[(2*l1+1)*(2*l2+1)],
                                      float (&tp)[51]) {
#pragma unroll
  for (int m = 0; m < 2*lo+1; ++m) {
    float acc = 0.0f;
#pragma unroll
    for (int a = 0; a < 2*l1+1; ++a)
#pragma unroll
      for (int b = 0; b < 2*l2+1; ++b)
        if (cg_nonzero(l1, l2, lo, a, b, m))
          acc = fmaf(mx[((KG+m)*9 + (IOFF+a))*9 + (JOFF+b)], p[a*(2*l2+1)+b], acc);
    tp[DST + m] = acc;
  }
}

// ---------------- WMMA fragment helpers ----------------
__device__ __forceinline__ v16bf combine16(v8bf lo, v8bf hi) {
  union { v16bf v; v8bf h[2]; } u;
  u.h[0] = lo; u.h[1] = hi;
  return u.v;
}

// One output-L GEMM slice for one 16-column tile `ct`:
//   C[rt (=m), :] += A(X_L rows rt*16..rt*16+15, K) x B(Wb_L cols ct*16..+15, K)
template<int RT, int KT, int STE, int XB, int OUTOFF>
__device__ __forceinline__ void gemm_part(const __bf16* sx,
                                          const __bf16* __restrict__ wbL,
                                          float* __restrict__ out,
                                          int zbase, int ct, int lane, float norm) {
  constexpr int K = KT * 32;
  const int l16 = lane & 15;
  const int lh  = lane >> 4;

  v8f acc[RT];
#pragma unroll
  for (int rt = 0; rt < RT; ++rt)
#pragma unroll
    for (int r = 0; r < 8; ++r) acc[rt][r] = 0.0f;

  // Rolled kt loop (full unroll caused fragment hoisting + scratch spills).
  // Within an iteration: issue all loads first (they clause & pipeline), then
  // the RT WMMAs, so ds latency overlaps matrix work.
#pragma unroll 1
  for (int kt = 0; kt < KT; ++kt) {
    // B fragment: 32(K) x 16(N) bf16. lane = column, 16 contiguous K per lane
    // (lanes 0-15: K kt*32..+15, lanes 16-31: K kt*32+16..+31)  [contiguous k, same as A]
    const __bf16* bp = wbL + (size_t)(ct*16 + l16) * K + kt*32 + lh*16;
    v16bf bfrag = combine16(*(const v8bf*)bp, *(const v8bf*)(bp + 8));

    // A fragments: 16(M) x 32(K) bf16. lane&15 = row, 16 contiguous K per lane:
    // lane<16: {k0..k0+15}, lane>=16: {k0+16..k0+31}
    v16bf af[RT];
#pragma unroll
    for (int rt = 0; rt < RT; ++rt) {
      const __bf16* ap = sx + XB + (size_t)(rt*16 + l16) * STE + kt*32 + lh*16;
      af[rt] = combine16(*(const v8bf*)ap, *(const v8bf*)(ap + 8));
    }

#pragma unroll
    for (int rt = 0; rt < RT; ++rt)
      acc[rt] = __builtin_amdgcn_wmma_f32_16x16x32_bf16(
          /*neg_a=*/false, af[rt], /*neg_b=*/false, bfrag,
          /*c_mod=*/(short)0, acc[rt], /*reuse_a=*/false, /*reuse_b=*/false);

    SCHED_FENCE();
  }

  // C layout: lanes 0-15: N=lane, M=r ; lanes 16-31: N=lane-16, M=r+8
  // Output is write-once (151 MB): non-temporal stores keep L2 for W/x streams.
#pragma unroll
  for (int rt = 0; rt < RT; ++rt)
#pragma unroll
    for (int r = 0; r < 8; ++r) {
      const int zloc = r + lh * 8;                       // row M = z_local
      const int col  = OUTOFF + (ct*16 + l16) * RT + rt; // out col = off + c*(2L+1)+m
      __builtin_nontemporal_store(acc[rt][r] * norm,
                                  out + (size_t)(zbase + zloc) * DIM_OUT + col);
    }
}

// ---------------- kernels ----------------
__global__ void __launch_bounds__(256)
convert_w_kernel(const float* __restrict__ w0, const float* __restrict__ w1,
                 const float* __restrict__ w2, __bf16* __restrict__ wb) {
  int i = blockIdx.x * 256 + threadIdx.x;
  if (i >= WB_TOTAL) return;
  float v;
  if (i < WB1_OFF)       v = w0[i];
  else if (i < WB2_OFF)  v = w1[i - WB1_OFF];
  else                   v = w2[i - WB2_OFF];
  wb[i] = (__bf16)v;
}

__global__ void __launch_bounds__(256, 4)   // 4 waves/SIMD target -> <=256 VGPRs, no spill
ltp_fused_kernel(const float* __restrict__ x1, const float* __restrict__ x2,
                 const float* __restrict__ mx, const __bf16* __restrict__ wb,
                 float* __restrict__ out) {
  extern __shared__ char smem_raw[];
  __bf16* sx = (__bf16*)smem_raw;

  const int t     = threadIdx.x;
  const int zbase = blockIdx.x * ZT;

  // ================= Stage 1: sparse CG tensor product (f32 VALU) =================
  // 256 threads x 4 pairs = 1024 (z_local, u) pairs per block.
#pragma unroll 1
  for (int r = 0; r < 4; ++r) {
    const int p  = (t << 2) | r;
    const int u  = p & 63;
    const int zl = p >> 6;

    float x1v[9], x2v[9];
    const float* p1 = x1 + (size_t)(zbase + zl) * (MUL_MID * DIM_MID) + u * DIM_MID;
    const float* p2 = x2 + (size_t)(zbase + zl) * (MUL_MID * DIM_MID) + u * DIM_MID;
#pragma unroll
    for (int i = 0; i < 9; ++i) { x1v[i] = p1[i]; x2v[i] = p2[i]; }

    // tp slots: [0..2]=L0 n ; [3..20]=L1 (3+3n+m) ; [21..50]=L2 (21+5n+m)
    float tp[51];

    { float pr[1];  tp_prod<0,0,0,0>(x1v,x2v,pr); tp_lo<0,0,0,0,0, 0, 0>(mx,pr,tp); }
    { float pr[3];  tp_prod<0,1,0,1>(x1v,x2v,pr); tp_lo<0,1,1,0,1, 1, 3>(mx,pr,tp); }
    { float pr[5];  tp_prod<0,2,0,4>(x1v,x2v,pr); tp_lo<0,2,2,0,4, 4,21>(mx,pr,tp); }
    { float pr[3];  tp_prod<1,0,1,0>(x1v,x2v,pr); tp_lo<1,0,1,1,0, 9, 6>(mx,pr,tp); }
    { float pr[9];  tp_prod<1,1,1,1>(x1v,x2v,pr);
      tp_lo<1,1,0,1,1,12, 1>(mx,pr,tp);
      tp_lo<1,1,1,1,1,13, 9>(mx,pr,tp);
      tp_lo<1,1,2,1,1,16,26>(mx,pr,tp); }
    { float pr[15]; tp_prod<1,2,1,4>(x1v,x2v,pr);
      tp_lo<1,2,1,1,4,21,12>(mx,pr,tp);
      tp_lo<1,2,2,1,4,24,31>(mx,pr,tp); }
    { float pr[5];  tp_prod<2,0,4,0>(x1v,x2v,pr); tp_lo<2,0,2,4,0,36,36>(mx,pr,tp); }
    { float pr[15]; tp_prod<2,1,4,1>(x1v,x2v,pr);
      tp_lo<2,1,1,4,1,41,15>(mx,pr,tp);
      tp_lo<2,1,2,4,1,44,41>(mx,pr,tp); }
    { float pr[25]; tp_prod<2,2,4,4>(x1v,x2v,pr);
      tp_lo<2,2,0,4,4,56, 2>(mx,pr,tp);
      tp_lo<2,2,1,4,4,57,18>(mx,pr,tp);
      tp_lo<2,2,2,4,4,60,46>(mx,pr,tp); }

    // Scatter into GEMM-ready bf16 X tiles in LDS: X_L[row = m*16+zl][k = u*NL+n]
#pragma unroll
    for (int n = 0; n < 3; ++n)
      sx[X0_BASE + zl * ST0 + u * 3 + n] = (__bf16)tp[n];
#pragma unroll
    for (int n = 0; n < 6; ++n)
#pragma unroll
      for (int m = 0; m < 3; ++m)
        sx[X1_BASE + (m * 16 + zl) * ST12 + u * 6 + n] = (__bf16)tp[3 + 3*n + m];
#pragma unroll
    for (int n = 0; n < 6; ++n)
#pragma unroll
      for (int m = 0; m < 5; ++m)
        sx[X2_BASE + (m * 16 + zl) * ST12 + u * 6 + n] = (__bf16)tp[21 + 5*n + m];
  }

  __syncthreads();

  // ================= Stage 2: equivariant linear = 3 GEMMs via bf16 WMMA =========
  const int lane = t & 31;
  const int wave = t >> 5;
  const __bf16* wb0 = wb + WB0_OFF;
  const __bf16* wb1 = wb + WB1_OFF;
  const __bf16* wb2 = wb + WB2_OFF;

#pragma unroll 1
  for (int cti = 0; cti < 2; ++cti) {
    const int ct = wave * 2 + cti;                       // 8 waves x 2 = 16 col tiles
    gemm_part<1,  6, ST0,  X0_BASE,    0>(sx, wb0, out, zbase, ct, lane, NORM_L0);
    gemm_part<3, 12, ST12, X1_BASE,  256>(sx, wb1, out, zbase, ct, lane, NORM_L12);
    gemm_part<5, 12, ST12, X2_BASE, 1024>(sx, wb2, out, zbase, ct, lane, NORM_L12);
  }
}

// ---------------- host entry ----------------
extern "C" void kernel_launch(void* const* d_in, const int* in_sizes, int n_in,
                              void* d_out, int out_size, void* d_ws, size_t ws_size,
                              hipStream_t stream) {
  const float* x1 = (const float*)d_in[0];
  const float* x2 = (const float*)d_in[1];
  const float* mx = (const float*)d_in[2];
  const float* w0 = (const float*)d_in[3];
  const float* w1 = (const float*)d_in[4];
  const float* w2 = (const float*)d_in[5];
  float* out = (float*)d_out;
  __bf16* wb = (__bf16*)d_ws;   // 491,520 B of scratch for bf16 weights

  // Stage 0: one-time (per call) W -> bf16 conversion into L2-resident staging.
  convert_w_kernel<<<(WB_TOTAL + 255) / 256, 256, 0, stream>>>(w0, w1, w2, wb);

  // Fused TP + WMMA-GEMM kernel: 1024 blocks x 256 threads, 111,360 B dynamic LDS.
  ltp_fused_kernel<<<Z / ZT, 256, LDS_BYTES, stream>>>(x1, x2, mx, wb, out);
}
